// BFeatVanillaGAT_90726889160929
// MI455X (gfx1250) — matrix operand order repaired and checked by reading, not compiled, not run on hardware
//
#include <hip/hip_runtime.h>
#include <hip/hip_bf16.h>
#include <math.h>

// ---------------- problem constants (from reference) ----------------
#define S_  16
#define L_  96
#define N_  (S_*L_)          // 1536 nodes
#define E_  (S_*L_*(L_-1))   // 145920 edges
#define D_  256
#define H_  8

typedef __bf16 bf16;
typedef __attribute__((ext_vector_type(16))) __bf16 bf16x16;
typedef __attribute__((ext_vector_type(8)))  float  f32x8;
typedef __attribute__((ext_vector_type(4)))  unsigned int u32x4;
typedef __attribute__((ext_vector_type(8)))  int i32x8;
typedef __attribute__((ext_vector_type(4)))  int i32x4;

__device__ __forceinline__ int lane_id() { return threadIdx.x & 31; }

__device__ __forceinline__ f32x8 wmma_bf16(bf16x16 a, bf16x16 b, f32x8 c) {
  return __builtin_amdgcn_wmma_f32_16x16x32_bf16(false, a, false, b, (short)0, c,
                                                 false, false);
}

// A fragment: 16x32 bf16 tile from row-major A (lane<16 holds
// K{0..15} of row=lane, lane>=16 holds K{16..31} of row=lane-16)
__device__ __forceinline__ bf16x16 load_a_frag(const bf16* A, int lda) {
  const int l  = lane_id();
  const int r  = l & 15;
  const int k0 = (l >> 4) * 16;
  const bf16* p0 = A + (long)r * lda + k0;
  const bf16* p1 = p0 + 8;
  bf16x16 f;
#pragma unroll
  for (int t = 0; t < 8; ++t) { f[t] = p0[t]; f[8 + t] = p1[t]; }
  return f;
}

// B fragment (32x16) where B[k][n] = W[n][k]; W row-major [Nout][K]
__device__ __forceinline__ bf16x16 load_b_fragT(const bf16* W, int ldw) {
  const int l  = lane_id();
  const int n  = l & 15;
  const int k0 = (l >> 4) * 16;
  const bf16* p = W + (long)n * ldw + k0;
  bf16x16 f;
#pragma unroll
  for (int t = 0; t < 16; ++t) f[t] = p[t];
  return f;
}

// B fragment from row-major B[K][N]
__device__ __forceinline__ bf16x16 load_b_fragRM(const bf16* B, int ldb) {
  const int l  = lane_id();
  const int n  = l & 15;
  const int k0 = (l >> 4) * 16;
  bf16x16 f;
#pragma unroll
  for (int t = 0; t < 16; ++t) f[t] = B[(long)(k0 + t) * ldb + n];
  return f;
}

__device__ __forceinline__ float wave_max(float v) {
#pragma unroll
  for (int o = 16; o > 0; o >>= 1) v = fmaxf(v, __shfl_xor(v, o, 32));
  return v;
}
__device__ __forceinline__ float wave_sum(float v) {
#pragma unroll
  for (int o = 16; o > 0; o >>= 1) v += __shfl_xor(v, o, 32);
  return v;
}

// TDM: async copy of `nelem` bf16 elements (flat tile) global -> LDS.
// D# per ISA ch8: group0 {count=1, lds_addr, global_addr, type=2},
// group1 {data_size=1(2B), tensor_dim0=tile_dim0=n, tile_dim1=1, stride=n}.
// clang-23 lane: 6-arg builtin (extra zero i32x8 before cpol).
__device__ __forceinline__ void tdm_load_flat(const bf16* gsrc, void* lds_dst,
                                              unsigned nelem) {
  unsigned ldsa = (unsigned)(uintptr_t)lds_dst;       // aperture low bits = LDS addr
  unsigned long long ga = (unsigned long long)(uintptr_t)gsrc;
  u32x4 g0 = { 1u,                                    // count=1, user desc
               ldsa,
               (unsigned)(ga & 0xffffffffu),
               (unsigned)((ga >> 32) & 0x01ffffffu) | 0x80000000u }; // type=2
  i32x8 g1 = { (int)0x00010000,                       // data_size=1 (2 bytes)
               (int)(nelem << 16),                    // tensor_dim0[15:0] @ [63:48]
               (int)(1u << 16),                       // tensor_dim1=1 @ [111:80]
               (int)(nelem << 16),                    // tile_dim0 @ [127:112]
               1,                                     // tile_dim1=1 @ [143:128]
               (int)nelem,                            // tensor_dim0_stride[31:0]
               0, 0 };
  i32x4 g2 = { 0, 0, 0, 0 }, g3 = { 0, 0, 0, 0 };
  i32x8 g4 = { 0, 0, 0, 0, 0, 0, 0, 0 };
  __builtin_amdgcn_tensor_load_to_lds(g0, g1, g2, g3, g4, 0);
}

// ---------------- fp32 -> bf16 convert ----------------
__global__ void k_cvt(const float* __restrict__ s, bf16* __restrict__ d, long n) {
  long i = (long)blockIdx.x * blockDim.x + threadIdx.x;
  if (i < n) d[i] = (bf16)s[i];
}

// ---------------- generic WMMA GEMM: C = act(A[M,K] * W[N,K]^T + b) ---------
// one wave per 16x64 output block: A fragment reused across 4 WMMAs
__global__ void __launch_bounds__(128)
k_gemm(const bf16* __restrict__ A, const bf16* __restrict__ W,
       const float* __restrict__ bias, float* __restrict__ Cf,
       bf16* __restrict__ Cb, int M, int Nd, int K, int relu) {
  const int tile = blockIdx.x * (blockDim.x >> 5) + (threadIdx.x >> 5);
  const int tn = Nd >> 6;                       // 64-wide column blocks
  if (tile >= (M >> 4) * tn) return;
  const int m0 = (tile / tn) << 4;
  const int n0 = (tile % tn) << 6;
  f32x8 acc[4] = {{}, {}, {}, {}};
  for (int kk = 0; kk < K; kk += 32) {
    bf16x16 a = load_a_frag(A + (long)m0 * K + kk, K);
#pragma unroll
    for (int t = 0; t < 4; ++t) {
      bf16x16 b = load_b_fragT(W + (long)(n0 + t * 16) * K + kk, K);
      acc[t] = wmma_bf16(a, b, acc[t]);
    }
  }
  const int l = lane_id();
#pragma unroll
  for (int t = 0; t < 4; ++t) {
    const int col = n0 + t * 16 + (l & 15);
    const float bv = bias ? bias[col] : 0.f;
#pragma unroll
    for (int i = 0; i < 8; ++i) {
      int row = m0 + ((l >> 4) << 3) + i;
      float v = acc[t][i] + bv;
      if (relu) v = fmaxf(v, 0.f);
      if (Cf) Cf[(long)row * Nd + col] = v;
      if (Cb) Cb[(long)row * Nd + col] = (bf16)v;
    }
  }
}

// ---------------- distance-bias MLP: per (scene,p,q) pair -------------------
__global__ void __launch_bounds__(256)
k_distbias(const float* __restrict__ center,
           const float* __restrict__ fc1w, const float* __restrict__ fc1b,
           const float* __restrict__ g1,   const float* __restrict__ b1,
           const float* __restrict__ fc2w, const float* __restrict__ fc2b,
           const float* __restrict__ g2,   const float* __restrict__ b2,
           const float* __restrict__ fc3w, const float* __restrict__ fc3b,
           float* __restrict__ dw) {
  __shared__ float s1w[32 * 4], s1b[32], sg1[32], sb1[32];
  __shared__ float s2w[32 * 32], s2b[32], sg2[32], sb2[32];
  __shared__ float s3w[8 * 32], s3b[8];
  for (int i = threadIdx.x; i < 128; i += blockDim.x) s1w[i] = fc1w[i];
  for (int i = threadIdx.x; i < 32; i += blockDim.x) {
    s1b[i] = fc1b[i]; sg1[i] = g1[i]; sb1[i] = b1[i];
    s2b[i] = fc2b[i]; sg2[i] = g2[i]; sb2[i] = b2[i];
  }
  for (int i = threadIdx.x; i < 1024; i += blockDim.x) s2w[i] = fc2w[i];
  for (int i = threadIdx.x; i < 256; i += blockDim.x) s3w[i] = fc3w[i];
  if (threadIdx.x < 8) s3b[threadIdx.x] = fc3b[threadIdx.x];
  __syncthreads();

  int idx = blockIdx.x * blockDim.x + threadIdx.x;
  if (idx >= S_ * L_ * L_) return;
  const int sc = idx / (L_ * L_);
  const int pq = idx % (L_ * L_);
  const int p = pq / L_, q = pq % L_;
  const float* cp = center + (long)(sc * L_ + p) * 3;
  const float* cq = center + (long)(sc * L_ + q) * 3;
  float w0 = cq[0] - cp[0], w1 = cq[1] - cp[1], w2 = cq[2] - cp[2];
  float w3 = sqrtf(w0 * w0 + w1 * w1 + w2 * w2);

  float h1a[32];
#pragma unroll
  for (int o = 0; o < 32; ++o) {
    float a = s1b[o] + w0 * s1w[o * 4] + w1 * s1w[o * 4 + 1] +
              w2 * s1w[o * 4 + 2] + w3 * s1w[o * 4 + 3];
    h1a[o] = fmaxf(a, 0.f);
  }
  float mu = 0.f;
#pragma unroll
  for (int o = 0; o < 32; ++o) mu += h1a[o];
  mu *= (1.f / 32.f);
  float var = 0.f;
#pragma unroll
  for (int o = 0; o < 32; ++o) { float d = h1a[o] - mu; var += d * d; }
  float inv = rsqrtf(var * (1.f / 32.f) + 1e-5f);
#pragma unroll
  for (int o = 0; o < 32; ++o) h1a[o] = (h1a[o] - mu) * inv * sg1[o] + sb1[o];

  float h2a[32];
  for (int o = 0; o < 32; ++o) {
    float a = s2b[o];
#pragma unroll
    for (int cix = 0; cix < 32; ++cix) a += h1a[cix] * s2w[o * 32 + cix];
    h2a[o] = fmaxf(a, 0.f);
  }
  mu = 0.f;
#pragma unroll
  for (int o = 0; o < 32; ++o) mu += h2a[o];
  mu *= (1.f / 32.f);
  var = 0.f;
#pragma unroll
  for (int o = 0; o < 32; ++o) { float d = h2a[o] - mu; var += d * d; }
  inv = rsqrtf(var * (1.f / 32.f) + 1e-5f);
#pragma unroll
  for (int o = 0; o < 32; ++o) h2a[o] = (h2a[o] - mu) * inv * sg2[o] + sb2[o];

  for (int hh = 0; hh < 8; ++hh) {
    float a = s3b[hh];
#pragma unroll
    for (int cix = 0; cix < 32; ++cix) a += h2a[cix] * s3w[hh * 32 + cix];
    dw[((long)(sc * 8 + hh) * L_ + p) * L_ + q] = a;   // same-scene mask == 1
  }
}

// ---------------- fused masked MHA per (scene, head) ------------------------
__global__ void __launch_bounds__(192)
k_attn(const bf16* __restrict__ qb, const bf16* __restrict__ kb,
       const bf16* __restrict__ vb, const float* __restrict__ dw,
       bf16* __restrict__ ob) {
  __shared__ float sc[6][16][96];
  __shared__ bf16  at[6][16][96];
  const int s = blockIdx.x >> 3;
  const int h = blockIdx.x & 7;
  const int w = threadIdx.x >> 5;
  const int l = lane_id();
  const float scale = 0.17677669529663687f;   // 1/sqrt(32)

  bf16x16 a = load_a_frag(qb + (long)(s * 96 + w * 16) * D_ + h * 32, D_);
  const float* dwp = dw + (long)(s * 8 + h) * 96 * 96;
#pragma unroll
  for (int nt = 0; nt < 6; ++nt) {
    bf16x16 b = load_b_fragT(kb + (long)(s * 96 + nt * 16) * D_ + h * 32, D_);
    f32x8 c = {};
    c = wmma_bf16(a, b, c);
    int col = nt * 16 + (l & 15);
#pragma unroll
    for (int i = 0; i < 8; ++i) {
      int row = ((l >> 4) << 3) + i;
      sc[w][row][col] = c[i] * scale + dwp[(w * 16 + row) * 96 + col];
    }
  }
  __syncthreads();
  if (l < 16) {                       // one lane per query row: softmax over 96
    float m = -3.4e38f;
    for (int j = 0; j < 96; ++j) m = fmaxf(m, sc[w][l][j]);
    float ssum = 0.f;
    for (int j = 0; j < 96; ++j) {
      float e = __expf(sc[w][l][j] - m);
      sc[w][l][j] = e; ssum += e;
    }
    float inv = 1.f / ssum;
    for (int j = 0; j < 96; ++j) at[w][l][j] = (bf16)(sc[w][l][j] * inv);
  }
  __syncthreads();
  // o = att(16x96) @ v(96x32): K = 3 chunks of 32, N = 2 tiles of 16
  f32x8 c0 = {}, c1 = {};
#pragma unroll
  for (int kc = 0; kc < 3; ++kc) {
    bf16x16 aa = load_a_frag((const bf16*)&at[w][0][0] + kc * 32, 96);
    bf16x16 b0 = load_b_fragRM(vb + (long)(s * 96 + kc * 32) * D_ + h * 32, D_);
    bf16x16 b1 = load_b_fragRM(vb + (long)(s * 96 + kc * 32) * D_ + h * 32 + 16, D_);
    c0 = wmma_bf16(aa, b0, c0);
    c1 = wmma_bf16(aa, b1, c1);
  }
  const int col = l & 15;
#pragma unroll
  for (int i = 0; i < 8; ++i) {
    int row = ((l >> 4) << 3) + i;
    long base = (long)(s * 96 + w * 16 + row) * D_ + h * 32;
    ob[base + col]      = (bf16)c0[i];
    ob[base + 16 + col] = (bf16)c1[i];
  }
}

// ---------------- residual + LayerNorm (wave per row) -----------------------
__global__ void __launch_bounds__(256)
k_resln(const float* __restrict__ obj, const float* __restrict__ t,
        const float* __restrict__ g, const float* __restrict__ b,
        float* __restrict__ xf, bf16* __restrict__ xb) {
  const int row = blockIdx.x * (blockDim.x >> 5) + (threadIdx.x >> 5);
  if (row >= N_) return;
  const int l = lane_id();
  float v[8];
  float s = 0.f;
#pragma unroll
  for (int t8 = 0; t8 < 8; ++t8) {
    int cix = l * 8 + t8;
    v[t8] = obj[(long)row * D_ + cix] + t[(long)row * D_ + cix];
    s += v[t8];
  }
  float mu = wave_sum(s) * (1.f / 256.f);
  float var = 0.f;
#pragma unroll
  for (int t8 = 0; t8 < 8; ++t8) { float d = v[t8] - mu; var += d * d; }
  float inv = rsqrtf(wave_sum(var) * (1.f / 256.f) + 1e-5f);
#pragma unroll
  for (int t8 = 0; t8 < 8; ++t8) {
    int cix = l * 8 + t8;
    float y = (v[t8] - mu) * inv * g[cix] + b[cix];
    xf[(long)row * D_ + cix] = y;
    xb[(long)row * D_ + cix] = (bf16)y;
  }
}

// ---------------- fused edge MLP: e_new = ne2(relu(ne1([xi|e|xj]))) ---------
// block = 8 waves, 16 edges. Contiguous edge tile comes in via TDM async DMA
// (overlapped with the scattered xi/xj VGPR gather); K-loop selects A source.
__global__ void __launch_bounds__(256)
k_edge_mlp(const bf16* __restrict__ xb, const bf16* __restrict__ eb,
           const int* __restrict__ eidx,
           const bf16* __restrict__ w1, const float* __restrict__ b1,
           const bf16* __restrict__ w2, const float* __restrict__ b2,
           float* __restrict__ outE) {
  __shared__ bf16 catA[16][512];    // [xi | xj]
  __shared__ bf16 eTile[16][256];   // TDM destination (contiguous edge rows)
  __shared__ bf16 h1[16][512];
  const int ebase = blockIdx.x * 16;
  const int tid = threadIdx.x;

  if (tid < 32) {                   // wave 0 kicks off the tensor DMA
    tdm_load_flat(eb + (long)ebase * D_, &eTile[0][0], 16 * D_);
  }
  for (int idx = tid; idx < 16 * 512; idx += 256) {   // gather xi / xj rows
    int r = idx >> 9, cix = idx & 511;
    int e = ebase + r;
    catA[r][cix] = (cix < 256) ? xb[(long)eidx[e] * D_ + cix]          // x[src]
                               : xb[(long)eidx[E_ + e] * D_ + (cix - 256)];
  }
  if (tid < 32) __builtin_amdgcn_s_wait_tensorcnt(0);
  __syncthreads();

  const int wv = tid >> 5;
  const int l = lane_id();
  // layer 1: 512 outputs, each wave owns 4 column tiles; A frag reused 4x
  {
    f32x8 acc[4] = {{}, {}, {}, {}};
    for (int kk = 0; kk < 768; kk += 32) {
      const bf16* ab; int lda;
      if (kk < 256)      { ab = &catA[0][kk];        lda = 512; }  // xi
      else if (kk < 512) { ab = &eTile[0][kk - 256]; lda = 256; }  // edge
      else               { ab = &catA[0][kk - 256];  lda = 512; }  // xj
      bf16x16 a = load_a_frag(ab, lda);
#pragma unroll
      for (int t = 0; t < 4; ++t) {
        int ct = wv * 4 + t;
        bf16x16 b = load_b_fragT(w1 + (long)(ct * 16) * 768 + kk, 768);
        acc[t] = wmma_bf16(a, b, acc[t]);
      }
    }
#pragma unroll
    for (int t = 0; t < 4; ++t) {
      int col = (wv * 4 + t) * 16 + (l & 15);
      float bias = b1[col];
#pragma unroll
      for (int i = 0; i < 8; ++i) {
        int row = ((l >> 4) << 3) + i;
        h1[row][col] = (bf16)fmaxf(acc[t][i] + bias, 0.f);
      }
    }
  }
  __syncthreads();
  // layer 2: 256 outputs, each wave owns 2 column tiles; A frag reused 2x
  {
    f32x8 acc[2] = {{}, {}};
    for (int kk = 0; kk < 512; kk += 32) {
      bf16x16 a = load_a_frag((const bf16*)&h1[0][0] + kk, 512);
#pragma unroll
      for (int t = 0; t < 2; ++t) {
        int ct = wv * 2 + t;
        bf16x16 b = load_b_fragT(w2 + (long)(ct * 16) * 512 + kk, 512);
        acc[t] = wmma_bf16(a, b, acc[t]);
      }
    }
#pragma unroll
    for (int t = 0; t < 2; ++t) {
      int col = (wv * 2 + t) * 16 + (l & 15);
      float bias = b2[col];
#pragma unroll
      for (int i = 0; i < 8; ++i) {
        int row = ((l >> 4) << 3) + i;
        outE[(long)(ebase + row) * D_ + col] = fmaxf(acc[t][i] + bias, 0.f);
      }
    }
  }
}

// ---------------- gcn: conv1d attention + softmax + segment-max -------------
__global__ void __launch_bounds__(256)
k_gcn(const float* __restrict__ qe, const float* __restrict__ val,
      const bf16* __restrict__ ee, const int* __restrict__ eidx,
      const float* __restrict__ c1w, const float* __restrict__ c1b,
      const float* __restrict__ c2w, const float* __restrict__ c2b,
      bf16* __restrict__ gcn) {
  __shared__ float s1w[64 * 64];
  __shared__ float s2w[32 * 64];
  __shared__ float s1b[64], s2b[32];
  __shared__ float qel[256];
  __shared__ float catl[8][64];
  __shared__ float hhl[8][64];
  const int i = blockIdx.x;
  const int tid = threadIdx.x;
  for (int k = tid; k < 64 * 64; k += 256) s1w[k] = c1w[k];
  for (int k = tid; k < 32 * 64; k += 256) s2w[k] = c2w[k];
  if (tid < 64) s1b[tid] = c1b[tid];
  if (tid < 32) s2b[tid] = c2b[tid];
  qel[tid] = qe[(long)i * D_ + tid];
  __syncthreads();

  const int h = tid >> 5, o = tid & 31;     // threads of same h == one wave32
  const int elem = o * 8 + h;               // prob.reshape(E,256) flat index
  float gmax = -3.4e38f;
  for (int t = 0; t < L_ - 1; ++t) {
    const long e = (long)i * (L_ - 1) + t;
    const int jn = eidx[E_ + e];
    for (int idx = tid; idx < 512; idx += 256) {
      int hh = idx >> 6, cix = idx & 63;
      catl[hh][cix] = (cix < 32) ? qel[cix * 8 + hh]
                                 : (float)ee[e * D_ + (cix - 32) * 8 + hh];
    }
    __syncthreads();
    for (int oi = tid; oi < 512; oi += 256) {       // c1: 64->64 per head, relu
      int hh = oi >> 6, oc = oi & 63;
      float acc = s1b[oc];
#pragma unroll 8
      for (int cix = 0; cix < 64; ++cix) acc += catl[hh][cix] * s1w[oc * 64 + cix];
      hhl[hh][oc] = fmaxf(acc, 0.f);
    }
    __syncthreads();
    float p = s2b[o];                               // c2: 64->32 per head
#pragma unroll 8
    for (int cix = 0; cix < 64; ++cix) p += hhl[h][cix] * s2w[o * 64 + cix];
    float m = wave_max(p);                          // softmax over channel dim
    float ex = __expf(p - m);
    float prob = ex / wave_sum(ex);
    float msg = prob * val[(long)jn * D_ + elem];
    gmax = fmaxf(gmax, msg);
    __syncthreads();
  }
  float g = (gmax > -3.0e38f) ? gmax : 0.f;         // empty segments -> 0
  gcn[(long)i * D_ + elem] = (bf16)g;
}

// ---------------- fused prop MLP: x_out = pr2(relu(pr1([x|gcn]))) -----------
__global__ void __launch_bounds__(256)
k_prop(const bf16* __restrict__ xb, const bf16* __restrict__ gb,
       const bf16* __restrict__ w1, const float* __restrict__ b1,
       const bf16* __restrict__ w2, const float* __restrict__ b2,
       float* __restrict__ outX) {
  __shared__ bf16 catA[16][512];
  __shared__ bf16 h1[16][512];
  const int rbase = blockIdx.x * 16;
  const int tid = threadIdx.x;
  for (int idx = tid; idx < 16 * 512; idx += 256) {
    int r = idx >> 9, cix = idx & 511;
    catA[r][cix] = (cix < 256) ? xb[(long)(rbase + r) * D_ + cix]
                               : gb[(long)(rbase + r) * D_ + (cix - 256)];
  }
  __syncthreads();
  const int wv = tid >> 5;
  const int l = lane_id();
  {
    f32x8 acc[4] = {{}, {}, {}, {}};
    for (int kk = 0; kk < 512; kk += 32) {
      bf16x16 a = load_a_frag((const bf16*)&catA[0][0] + kk, 512);
#pragma unroll
      for (int t = 0; t < 4; ++t) {
        int ct = wv * 4 + t;
        bf16x16 b = load_b_fragT(w1 + (long)(ct * 16) * 512 + kk, 512);
        acc[t] = wmma_bf16(a, b, acc[t]);
      }
    }
#pragma unroll
    for (int t = 0; t < 4; ++t) {
      int col = (wv * 4 + t) * 16 + (l & 15);
      float bias = b1[col];
#pragma unroll
      for (int i = 0; i < 8; ++i) {
        int row = ((l >> 4) << 3) + i;
        h1[row][col] = (bf16)fmaxf(acc[t][i] + bias, 0.f);
      }
    }
  }
  __syncthreads();
  {
    f32x8 acc[2] = {{}, {}};
    for (int kk = 0; kk < 512; kk += 32) {
      bf16x16 a = load_a_frag((const bf16*)&h1[0][0] + kk, 512);
#pragma unroll
      for (int t = 0; t < 2; ++t) {
        int ct = wv * 2 + t;
        bf16x16 b = load_b_fragT(w2 + (long)(ct * 16) * 512 + kk, 512);
        acc[t] = wmma_bf16(a, b, acc[t]);
      }
    }
#pragma unroll
    for (int t = 0; t < 2; ++t) {
      int col = (wv * 2 + t) * 16 + (l & 15);
      float bias = b2[col];
#pragma unroll
      for (int i = 0; i < 8; ++i) {
        int row = ((l >> 4) << 3) + i;
        outX[(long)(rbase + row) * D_ + col] = fmaxf(acc[t][i] + bias, 0.f);
      }
    }
  }
}

// ---------------- host orchestration ----------------------------------------
extern "C" void kernel_launch(void* const* d_in, const int* in_sizes, int n_in,
                              void* d_out, int out_size, void* d_ws, size_t ws_size,
                              hipStream_t stream) {
  (void)in_sizes; (void)n_in; (void)out_size; (void)ws_size;
  const float* obj    = (const float*)d_in[0];
  const float* edge   = (const float*)d_in[1];
  const int*   eidx   = (const int*)d_in[2];
  const float* center = (const float*)d_in[4];
  int pi = 5;
  const float* fc1w = (const float*)d_in[pi++]; const float* fc1b = (const float*)d_in[pi++];
  const float* ln1g = (const float*)d_in[pi++]; const float* ln1b = (const float*)d_in[pi++];
  const float* fc2w = (const float*)d_in[pi++]; const float* fc2b = (const float*)d_in[pi++];
  const float* ln2g = (const float*)d_in[pi++]; const float* ln2b = (const float*)d_in[pi++];
  const float* fc3w = (const float*)d_in[pi++]; const float* fc3b = (const float*)d_in[pi++];
  const float* wqw  = (const float*)d_in[pi++]; const float* wqb  = (const float*)d_in[pi++];
  const float* wkw  = (const float*)d_in[pi++]; const float* wkb  = (const float*)d_in[pi++];
  const float* wvw  = (const float*)d_in[pi++]; const float* wvb  = (const float*)d_in[pi++];
  const float* wow  = (const float*)d_in[pi++]; const float* wob  = (const float*)d_in[pi++];
  const float* mlg  = (const float*)d_in[pi++]; const float* mlb  = (const float*)d_in[pi++];
  const float* ne1w = (const float*)d_in[pi++]; const float* ne1b = (const float*)d_in[pi++];
  const float* ne2w = (const float*)d_in[pi++]; const float* ne2b = (const float*)d_in[pi++];
  const float* pqw  = (const float*)d_in[pi++]; const float* pqb  = (const float*)d_in[pi++];
  const float* pew  = (const float*)d_in[pi++]; const float* peb  = (const float*)d_in[pi++];
  const float* pvw  = (const float*)d_in[pi++]; const float* pvb  = (const float*)d_in[pi++];
  const float* c1w  = (const float*)d_in[pi++]; const float* c1b  = (const float*)d_in[pi++];
  const float* c2w  = (const float*)d_in[pi++]; const float* c2b  = (const float*)d_in[pi++];
  const float* pr1w = (const float*)d_in[pi++]; const float* pr1b = (const float*)d_in[pi++];
  const float* pr2w = (const float*)d_in[pi++]; const float* pr2b = (const float*)d_in[pi++];

  float* out_x = (float*)d_out;                 // [N,256] relu(x_out)
  float* out_e = out_x + (long)N_ * D_;         // [E,256] relu(e_new)

  char* wp = (char*)d_ws;
  auto alloc = [&](size_t bytes) -> void* {
    void* p = (void*)wp; wp += (bytes + 255) & ~(size_t)255; return p;
  };
  float* dw_b    = (float*)alloc(sizeof(float) * S_ * H_ * L_ * L_);
  float* tmp_o   = (float*)alloc(sizeof(float) * N_ * D_);
  float* xf      = (float*)alloc(sizeof(float) * N_ * D_);
  float* qe_f    = (float*)alloc(sizeof(float) * N_ * D_);
  float* val_f   = (float*)alloc(sizeof(float) * N_ * D_);
  bf16* obj_b  = (bf16*)alloc(sizeof(bf16) * N_ * D_);
  bf16* q_b    = (bf16*)alloc(sizeof(bf16) * N_ * D_);
  bf16* k_b    = (bf16*)alloc(sizeof(bf16) * N_ * D_);
  bf16* v_b    = (bf16*)alloc(sizeof(bf16) * N_ * D_);
  bf16* o_b    = (bf16*)alloc(sizeof(bf16) * N_ * D_);
  bf16* x_b    = (bf16*)alloc(sizeof(bf16) * N_ * D_);
  bf16* gcn_b  = (bf16*)alloc(sizeof(bf16) * N_ * D_);
  bf16* edge_b = (bf16*)alloc(sizeof(bf16) * (size_t)E_ * D_);
  bf16* ee_b   = (bf16*)alloc(sizeof(bf16) * (size_t)E_ * D_);
  bf16* wq_bf  = (bf16*)alloc(sizeof(bf16) * 256 * 256);
  bf16* wk_bf  = (bf16*)alloc(sizeof(bf16) * 256 * 256);
  bf16* wv_bf  = (bf16*)alloc(sizeof(bf16) * 256 * 256);
  bf16* wo_bf  = (bf16*)alloc(sizeof(bf16) * 256 * 256);
  bf16* ne1_bf = (bf16*)alloc(sizeof(bf16) * 512 * 768);
  bf16* ne2_bf = (bf16*)alloc(sizeof(bf16) * 256 * 512);
  bf16* pq_bf  = (bf16*)alloc(sizeof(bf16) * 256 * 256);
  bf16* pe_bf  = (bf16*)alloc(sizeof(bf16) * 256 * 256);
  bf16* pv_bf  = (bf16*)alloc(sizeof(bf16) * 256 * 256);
  bf16* pr1_bf = (bf16*)alloc(sizeof(bf16) * 512 * 512);
  bf16* pr2_bf = (bf16*)alloc(sizeof(bf16) * 256 * 512);

  auto cvt = [&](const float* s, bf16* d, long n) {
    k_cvt<<<(unsigned)((n + 255) / 256), 256, 0, stream>>>(s, d, n);
  };
  cvt(obj, obj_b, (long)N_ * D_);
  cvt(edge, edge_b, (long)E_ * D_);
  cvt(wqw, wq_bf, 65536); cvt(wkw, wk_bf, 65536); cvt(wvw, wv_bf, 65536);
  cvt(wow, wo_bf, 65536);
  cvt(ne1w, ne1_bf, 512 * 768); cvt(ne2w, ne2_bf, 256 * 512);
  cvt(pqw, pq_bf, 65536); cvt(pew, pe_bf, 65536); cvt(pvw, pv_bf, 65536);
  cvt(pr1w, pr1_bf, 512 * 512); cvt(pr2w, pr2_bf, 256 * 512);

  auto gemm = [&](const bf16* A, const bf16* W, const float* bias, float* Cf,
                  bf16* Cb, int M, int Nd, int K, int relu) {
    int tiles = (M / 16) * (Nd / 64);
    k_gemm<<<(tiles + 3) / 4, 128, 0, stream>>>(A, W, bias, Cf, Cb, M, Nd, K, relu);
  };

  // distance-bias MLP -> dw[s][h][96][96]
  k_distbias<<<(S_ * L_ * L_ + 255) / 256, 256, 0, stream>>>(
      center, fc1w, fc1b, ln1g, ln1b, fc2w, fc2b, ln2g, ln2b, fc3w, fc3b, dw_b);

  // QKV projections (bf16 outputs for attention WMMA)
  gemm(obj_b, wq_bf, wqb, nullptr, q_b, N_, D_, D_, 0);
  gemm(obj_b, wk_bf, wkb, nullptr, k_b, N_, D_, D_, 0);
  gemm(obj_b, wv_bf, wvb, nullptr, v_b, N_, D_, D_, 0);

  // fused attention per (scene, head)
  k_attn<<<S_ * H_, 192, 0, stream>>>(q_b, k_b, v_b, dw_b, o_b);

  // output projection + residual LayerNorm -> x (f32 + bf16)
  gemm(o_b, wo_bf, wob, tmp_o, nullptr, N_, D_, D_, 0);
  k_resln<<<N_ / 8, 256, 0, stream>>>(obj, tmp_o, mlg, mlb, xf, x_b);

  // per-node / per-edge projections
  gemm(x_b, pq_bf, pqb, qe_f, nullptr, N_, D_, D_, 0);
  gemm(x_b, pv_bf, pvb, val_f, nullptr, N_, D_, D_, 0);
  gemm(edge_b, pe_bf, peb, nullptr, ee_b, E_, D_, D_, 0);

  // fused two-layer edge MLP (TDM-staged edge tile) -> relu(e_new) in d_out
  k_edge_mlp<<<E_ / 16, 256, 0, stream>>>(x_b, edge_b, eidx, ne1_bf, ne1b,
                                          ne2_bf, ne2b, out_e);

  // conv1d attention + channel softmax + segment max -> gcn features
  k_gcn<<<N_, 256, 0, stream>>>(qe_f, val_f, ee_b, eidx, c1w, c1b, c2w, c2b, gcn_b);

  // fused prop MLP -> relu(x_out) into d_out
  k_prop<<<N_ / 16, 256, 0, stream>>>(x_b, gcn_b, pr1_bf, pr1b, pr2_bf, pr2b, out_x);
}
